// ASRDecoder_57767310131297
// MI455X (gfx1250) — matrix operand order repaired and checked by reading, not compiled, not run on hardware
//
#include <hip/hip_runtime.h>
#include <cstdint>
#include <cstddef>

// ---------------------------------------------------------------------------
// ASR decoder for MI455X (gfx1250, wave32, WMMA + async-to-LDS staging).
//
//  * mean-pool commutes with the linear projection -> 50 GFLOP becomes 0.1.
//  * one-hot LSTM inputs -> x @ W_ih0.T is a row gather of W_ih0^T.
//  * all GEMMs on v_wmma_f32_16x16x32_bf16 (bf16 in, f32 accumulate).
//  * LSTM step kernels stage A/B k-blocks into LDS with
//    GLOBAL_LOAD_ASYNC_TO_LDS_B128 (ASYNCcnt DMA), double buffered:
//    3x less L2 traffic than per-wave direct loads.
// ---------------------------------------------------------------------------

#define BB     64
#define TAUD   500
#define DIN    768
#define HH     1024
#define G4     4096
#define SEQ    128
#define NSTEPS 127   // xs = [audio_enc] ++ one_hot(text[:,1:127]) -> 127 steps

typedef __bf16 bf16_t;
typedef __bf16 v16bf __attribute__((ext_vector_type(16)));
typedef __bf16 v8bf  __attribute__((ext_vector_type(8)));
typedef float  v8f   __attribute__((ext_vector_type(8)));

__device__ __forceinline__ float sigmoidf_(float x) {
  return 1.0f / (1.0f + __expf(-x));
}

// ---- CDNA5 async copy: global -> LDS, 16B per lane, tracked by ASYNCcnt ----
__device__ __forceinline__ void async_load_b128(uint32_t lds_off, uint64_t gaddr) {
  asm volatile("global_load_async_to_lds_b128 %0, %1, off"
               :
               : "v"(lds_off), "v"(gaddr)
               : "memory");
}
__device__ __forceinline__ void wait_asynccnt0() {
  asm volatile("s_wait_asynccnt 0x0" ::: "memory");
}

// ---- WMMA operand loaders (bf16 16x16x32, contiguous k order, same for A and B) ----
// A (16x32, MxK), row-major A[lda]:
//   lanes 0-15 : M=lane,   elems[e]=K(k0+e)
//   lanes 16-31: M=lane-16,elems[e]=K(k0+16+e)
__device__ __forceinline__ v16bf load_a_tile(const bf16_t* __restrict__ A, int lda,
                                             int m0, int k0, int lane) {
  int l = lane & 15, hi = lane >> 4;
  return *(const v16bf*)(A + (size_t)(m0 + l) * lda + k0 + hi * 16);
}

// B (32x16, KxN) with B[k][n] = W[n][k], W row-major [N x ldb]:
//   lanes 0-15 : N=n0+lane,    elems[e] = W[n][k0+e]      (32 contiguous bytes)
//   lanes 16-31: N=n0+lane-16, elems[e] = W[n][k0+16+e]
__device__ __forceinline__ v16bf load_b_tile(const bf16_t* __restrict__ W, int ldb,
                                             int n0, int k0, int lane) {
  int l = lane & 15, hi = lane >> 4;
  return *(const v16bf*)(W + (size_t)(n0 + l) * ldb + k0 + hi * 16);
}

__device__ __forceinline__ v8f wmma_bf16(v16bf a, v16bf b, v8f c) {
  return __builtin_amdgcn_wmma_f32_16x16x32_bf16(false, a, false, b, (short)0, c,
                                                 false, false);
}

// ---------------------------------------------------------------------------
// Generic bf16 GEMM (direct loads): out[M,N] = A[M,K] @ W[N,K]^T (+ bias),
// M = 64*gridDim.y. Used for setup + the big parallel output GEMM.
// ---------------------------------------------------------------------------
template <bool BIAS, bool OUT_BF16>
__global__ void gemm_bf16_kernel(const bf16_t* __restrict__ A,
                                 const bf16_t* __restrict__ W,
                                 const float* __restrict__ bias,
                                 void* __restrict__ out, int K, int N) {
  int lane = threadIdx.x & 31, w = threadIdx.x >> 5;
  int m0 = blockIdx.y * 64 + 16 * (w & 3);
  int n0 = blockIdx.x * 64 + 32 * (w >> 2);
  v8f acc0 = {}, acc1 = {};
  for (int k0 = 0; k0 < K; k0 += 32) {
    __builtin_prefetch(W + (size_t)(n0 + (lane & 15)) * K + k0 + 128, 0, 1);
    v16bf a  = load_a_tile(A, K, m0, k0, lane);
    v16bf b0 = load_b_tile(W, K, n0, k0, lane);
    v16bf b1 = load_b_tile(W, K, n0 + 16, k0, lane);
    acc0 = wmma_bf16(a, b0, acc0);
    acc1 = wmma_bf16(a, b1, acc1);
  }
  int l = lane & 15, hi = lane >> 4;
#pragma unroll
  for (int r = 0; r < 8; ++r) {
    int m = m0 + hi * 8 + r;
    int c0i = n0 + l, c1i = n0 + 16 + l;
    float v0 = acc0[r], v1 = acc1[r];
    if (BIAS) { v0 += bias[c0i]; v1 += bias[c1i]; }
    if (OUT_BF16) {
      ((bf16_t*)out)[(size_t)m * N + c0i] = (bf16_t)v0;
      ((bf16_t*)out)[(size_t)m * N + c1i] = (bf16_t)v1;
    } else {
      ((float*)out)[(size_t)m * N + c0i] = v0;
      ((float*)out)[(size_t)m * N + c1i] = v1;
    }
  }
}

// ---------------------------------------------------------------------------
// LDS-staged GEMM pass for the LSTM recurrence.
// Block = 256 thr (8 waves); block owns hidden cols [nb, nb+16) of all 4 gate
// panels; wave w: gate g=w&3, batch rows [32*(w>>2), +32).
// K is chopped into 16 blocks of 64; each block's A (64x64, 8KB) and B panel
// (4 gates x 16 rows x 64, 8KB) are DMA'd to LDS via async-to-LDS (4 b128
// per thread), double-buffered, one wait+barrier per block.
// ---------------------------------------------------------------------------
#define KBLK   64
#define NKB    (HH / KBLK)            // 16
#define STG_SZ 16384                  // 8KB A + 8KB B per buffer

__device__ __forceinline__ void stage_kblock(const bf16_t* __restrict__ Asrc,
                                             const bf16_t* __restrict__ Wsrc,
                                             int nb, int k0, uint32_t lds_dst,
                                             int tid) {
  // A: 64 rows x 64 cols bf16 -> LDS [row][128B]; 8 x 16B chunks per row
#pragma unroll
  for (int j = 0; j < 2; ++j) {
    int ch = j * 256 + tid;           // 0..511
    int r = ch >> 3, cb = ch & 7;
    uint64_t g = (uint64_t)(uintptr_t)(Asrc + (size_t)r * HH + k0) + cb * 16;
    async_load_b128(lds_dst + r * 128 + cb * 16, g);
  }
  // B: row (gate*16+nr) <- W row (gate*1024 + nb + nr), same 64-col slice
#pragma unroll
  for (int j = 0; j < 2; ++j) {
    int ch = j * 256 + tid;
    int r = ch >> 3, cb = ch & 7;
    int gate = r >> 4, nr = r & 15;
    uint64_t g = (uint64_t)(uintptr_t)(Wsrc + ((size_t)gate * HH + nb + nr) * HH + k0)
                 + cb * 16;
    async_load_b128(lds_dst + 8192 + r * 128 + cb * 16, g);
  }
}

__device__ __forceinline__ void gemm_pass_lds(const bf16_t* __restrict__ Asrc,
                                              const bf16_t* __restrict__ Wsrc,
                                              int nb, char* smem,
                                              v8f& acc0, v8f& acc1) {
  int tid = threadIdx.x;
  int lane = tid & 31, w = tid >> 5;
  int g = w & 3, mhalf = w >> 2;
  int m0a = mhalf * 32, m0b = m0a + 16;
  uint32_t lds0 = (uint32_t)(uintptr_t)smem;

  stage_kblock(Asrc, Wsrc, nb, 0, lds0, tid);       // prologue: block 0 -> buf0
  for (int kb = 0; kb < NKB; ++kb) {
    int buf = kb & 1;
    wait_asynccnt0();       // my DMAs into `buf` landed
    __syncthreads();        // everyone's did; also fences last block's reads
    if (kb + 1 < NKB)
      stage_kblock(Asrc, Wsrc, nb, (kb + 1) * KBLK, lds0 + (buf ^ 1) * STG_SZ, tid);
    const bf16_t* As = (const bf16_t*)(smem + buf * STG_SZ);
    const bf16_t* Bs = (const bf16_t*)(smem + buf * STG_SZ + 8192);
#pragma unroll
    for (int kc = 0; kc < KBLK; kc += 32) {
      v16bf b  = load_b_tile(Bs, KBLK, g * 16, kc, lane);
      v16bf a0 = load_a_tile(As, KBLK, m0a, kc, lane);
      v16bf a1 = load_a_tile(As, KBLK, m0b, kc, lane);
      acc0 = wmma_bf16(a0, b, acc0);
      acc1 = wmma_bf16(a1, b, acc1);
    }
  }
}

// ---------------------------------------------------------------------------
// Fused LSTM layer-0 step: gates = h_prev @ W_hh0^T + x_contrib + biases.
//   x_contrib: t==0 -> precomputed audio_enc@W_ih0^T; t>=1 -> one-hot gather
//   from pre-transposed W_ih0T[1024,4096]. Gate exchange via LDS (aliases
//   stage buffer 0), then sigmoid/tanh cell -> bf16 ping-pong h buffer.
// ---------------------------------------------------------------------------
__global__ void lstm_l0_step_kernel(const bf16_t* __restrict__ hprev,
                                    bf16_t* __restrict__ hnext,
                                    float* __restrict__ c,
                                    const bf16_t* __restrict__ Whh,
                                    const float* __restrict__ WihT,
                                    const float* __restrict__ x0buf,
                                    const int* __restrict__ text, int t,
                                    const float* __restrict__ bih,
                                    const float* __restrict__ bhh) {
  __shared__ __align__(16) char smem[2 * STG_SZ];   // 32KB stage; gl aliases buf0
  float* gl = (float*)smem;                         // [4][64][16] = 16KB
  int lane = threadIdx.x & 31, w = threadIdx.x >> 5;
  int g = w & 3, mhalf = w >> 2;
  int nb = blockIdx.x * 16;
  int m0a = mhalf * 32, m0b = m0a + 16;
  v8f acc0 = {}, acc1 = {};
  gemm_pass_lds(hprev, Whh, nb, smem, acc0, acc1);
  __syncthreads();                                  // stage reads done -> reuse as gl

  int l = lane & 15, hi = lane >> 4;
  int jg = g * HH + nb + l;                         // global gate index [0,4096)
  float bsum = bih[jg] + bhh[jg];
#pragma unroll
  for (int r = 0; r < 8; ++r) {
    int ma = m0a + hi * 8 + r, mb = m0b + hi * 8 + r;
    float xa, xb;
    if (t == 0) {
      xa = x0buf[(size_t)ma * G4 + jg];
      xb = x0buf[(size_t)mb * G4 + jg];
    } else {
      xa = WihT[(size_t)text[ma * SEQ + t] * G4 + jg];
      xb = WihT[(size_t)text[mb * SEQ + t] * G4 + jg];
    }
    gl[(g * 64 + ma) * 16 + l] = acc0[r] + bsum + xa;
    gl[(g * 64 + mb) * 16 + l] = acc1[r] + bsum + xb;
  }
  __syncthreads();
  for (int e = threadIdx.x; e < 1024; e += 256) {
    int m = e >> 4, ln = e & 15;
    float gi = gl[(0 * 64 + m) * 16 + ln], gf = gl[(1 * 64 + m) * 16 + ln];
    float gg = gl[(2 * 64 + m) * 16 + ln], go = gl[(3 * 64 + m) * 16 + ln];
    size_t idx = (size_t)m * HH + nb + ln;
    float cc = sigmoidf_(gf) * c[idx] + sigmoidf_(gi) * tanhf(gg);
    c[idx] = cc;
    hnext[idx] = (bf16_t)(sigmoidf_(go) * tanhf(cc));
  }
}

// ---------------------------------------------------------------------------
// Fused LSTM layer-1 step: gates = x_in @ W_ih1^T + h_prev @ W_hh1^T + biases
// (two accumulated LDS-staged passes), cell, writes h1 (ping-pong) and hs[t].
// ---------------------------------------------------------------------------
__global__ void lstm_l1_step_kernel(const bf16_t* __restrict__ xin,
                                    const bf16_t* __restrict__ hprev,
                                    bf16_t* __restrict__ hnext,
                                    float* __restrict__ c,
                                    const bf16_t* __restrict__ Wih,
                                    const bf16_t* __restrict__ Whh,
                                    const float* __restrict__ bih,
                                    const float* __restrict__ bhh,
                                    bf16_t* __restrict__ hs_t) {
  __shared__ __align__(16) char smem[2 * STG_SZ];
  float* gl = (float*)smem;
  int lane = threadIdx.x & 31, w = threadIdx.x >> 5;
  int g = w & 3, mhalf = w >> 2;
  int nb = blockIdx.x * 16;
  int m0a = mhalf * 32, m0b = m0a + 16;
  v8f acc0 = {}, acc1 = {};
  gemm_pass_lds(xin, Wih, nb, smem, acc0, acc1);
  gemm_pass_lds(hprev, Whh, nb, smem, acc0, acc1);
  __syncthreads();

  int l = lane & 15, hi = lane >> 4;
  int jg = g * HH + nb + l;
  float bsum = bih[jg] + bhh[jg];
#pragma unroll
  for (int r = 0; r < 8; ++r) {
    int ma = m0a + hi * 8 + r, mb = m0b + hi * 8 + r;
    gl[(g * 64 + ma) * 16 + l] = acc0[r] + bsum;
    gl[(g * 64 + mb) * 16 + l] = acc1[r] + bsum;
  }
  __syncthreads();
  for (int e = threadIdx.x; e < 1024; e += 256) {
    int m = e >> 4, ln = e & 15;
    float gi = gl[(0 * 64 + m) * 16 + ln], gf = gl[(1 * 64 + m) * 16 + ln];
    float gg = gl[(2 * 64 + m) * 16 + ln], go = gl[(3 * 64 + m) * 16 + ln];
    size_t idx = (size_t)m * HH + nb + ln;
    float cc = sigmoidf_(gf) * c[idx] + sigmoidf_(gi) * tanhf(gg);
    c[idx] = cc;
    bf16_t hb = (bf16_t)(sigmoidf_(go) * tanhf(cc));
    hnext[idx] = hb;
    hs_t[idx] = hb;
  }
}

// ---- small setup kernels ---------------------------------------------------

__global__ void audio_mean_kernel(const float* __restrict__ audio,
                                  bf16_t* __restrict__ out) {
  int i = blockIdx.x * 256 + threadIdx.x;  // 64*768 threads
  int b = i / DIN, d = i - b * DIN;
  const float* p = audio + (size_t)b * TAUD * DIN + d;
  float s = 0.f;
  for (int t = 0; t < TAUD; ++t) s += p[(size_t)t * DIN];
  out[i] = (bf16_t)(s * (1.0f / (float)TAUD));
}

__global__ void f32_to_bf16_kernel(const float* __restrict__ in,
                                   bf16_t* __restrict__ out, int n) {
  int i = blockIdx.x * 256 + threadIdx.x;
  if (i < n) out[i] = (bf16_t)in[i];
}

// W_ih0 [4096,1024] -> W_ih0T [1024,4096] (fp32, LDS-tiled)
__global__ void transpose_f32_kernel(const float* __restrict__ in,
                                     float* __restrict__ out) {
  __shared__ float tile[32][33];
  int c0 = blockIdx.x * 32;  // k  (0..1023)
  int r0 = blockIdx.y * 32;  // j  (0..4095)
  int tx = threadIdx.x, ty = threadIdx.y;  // 32x8
#pragma unroll
  for (int i = 0; i < 32; i += 8)
    tile[ty + i][tx] = in[(size_t)(r0 + ty + i) * HH + c0 + tx];
  __syncthreads();
#pragma unroll
  for (int i = 0; i < 32; i += 8)
    out[(size_t)(c0 + ty + i) * G4 + r0 + tx] = tile[tx][ty + i];
}

__global__ void zero_u4_kernel(uint4* __restrict__ p, int n) {
  int i = blockIdx.x * 256 + threadIdx.x;
  if (i < n) p[i] = make_uint4(0u, 0u, 0u, 0u);
}

// ---------------------------------------------------------------------------
extern "C" void kernel_launch(void* const* d_in, const int* in_sizes, int n_in,
                              void* d_out, int out_size, void* d_ws, size_t ws_size,
                              hipStream_t stream) {
  (void)in_sizes; (void)n_in; (void)out_size; (void)ws_size;
  const float* audio  = (const float*)d_in[0];
  const int*   text   = (const int*)d_in[1];
  const float* W_proj = (const float*)d_in[2];
  const float* b_proj = (const float*)d_in[3];
  const float* W_ih0  = (const float*)d_in[4];
  const float* W_hh0  = (const float*)d_in[5];
  const float* b_ih0  = (const float*)d_in[6];
  const float* b_hh0  = (const float*)d_in[7];
  const float* W_ih1  = (const float*)d_in[8];
  const float* W_hh1  = (const float*)d_in[9];
  const float* b_ih1  = (const float*)d_in[10];
  const float* b_hh1  = (const float*)d_in[11];
  const float* W_out  = (const float*)d_in[12];
  const float* b_out  = (const float*)d_in[13];
  float* logits = (float*)d_out;  // [127,64,1024] (t-major)

  uint8_t* ws = (uint8_t*)d_ws;
  size_t o = 0;
  auto take = [&](size_t bytes) -> uint8_t* {
    uint8_t* p = ws + o;
    o = (o + bytes + 255) & ~(size_t)255;
    return p;
  };
  bf16_t* Wb_hh0  = (bf16_t*)take((size_t)G4 * HH * 2);
  bf16_t* Wb_ih1  = (bf16_t*)take((size_t)G4 * HH * 2);
  bf16_t* Wb_hh1  = (bf16_t*)take((size_t)G4 * HH * 2);
  bf16_t* Wb_ih0  = (bf16_t*)take((size_t)G4 * HH * 2);
  bf16_t* Wb_out  = (bf16_t*)take((size_t)HH * HH * 2);
  bf16_t* Wb_proj = (bf16_t*)take((size_t)HH * DIN * 2);
  float*  W_ih0T  = (float*)take((size_t)HH * G4 * 4);    // [1024,4096]
  bf16_t* amean   = (bf16_t*)take((size_t)BB * DIN * 2);  // [64,768]
  bf16_t* aenc    = (bf16_t*)take((size_t)BB * HH * 2);   // [64,1024]
  float*  x0buf   = (float*)take((size_t)BB * G4 * 4);    // [64,4096]
  uint8_t* state  = take(4 * (size_t)BB * HH * 4);        // 1 MB, zeroed
  bf16_t* h0 = (bf16_t*)state;                            // 2 ping-pong bufs
  bf16_t* h1 = (bf16_t*)(state + 2 * (size_t)BB * HH * 2);
  float*  c0 = (float*)(state + 4 * (size_t)BB * HH * 2);
  float*  c1 = (float*)(state + 4 * (size_t)BB * HH * 2 + (size_t)BB * HH * 4);
  bf16_t* hs = (bf16_t*)take((size_t)NSTEPS * BB * HH * 2);  // [127,64,1024]

  // 1) setup: mean-pool, weight converts, transpose, zero state
  audio_mean_kernel<<<(BB * DIN) / 256, 256, 0, stream>>>(audio, amean);
  auto cvt = [&](const float* src, bf16_t* dst, int n) {
    f32_to_bf16_kernel<<<(n + 255) / 256, 256, 0, stream>>>(src, dst, n);
  };
  cvt(W_proj, Wb_proj, HH * DIN);
  cvt(W_ih0, Wb_ih0, G4 * HH);
  cvt(W_hh0, Wb_hh0, G4 * HH);
  cvt(W_ih1, Wb_ih1, G4 * HH);
  cvt(W_hh1, Wb_hh1, G4 * HH);
  cvt(W_out, Wb_out, HH * HH);
  transpose_f32_kernel<<<dim3(HH / 32, G4 / 32), dim3(32, 8), 0, stream>>>(W_ih0, W_ih0T);
  zero_u4_kernel<<<(4 * BB * HH * 4 / 16 + 255) / 256, 256, 0, stream>>>(
      (uint4*)state, 4 * BB * HH * 4 / 16);

  // 2) audio_enc = mean(audio) @ W_proj^T + b_proj  (bf16 out)
  gemm_bf16_kernel<true, true><<<dim3(HH / 64, 1), 256, 0, stream>>>(
      amean, Wb_proj, b_proj, aenc, DIN, HH);
  // 3) x0 = audio_enc @ W_ih0^T  (step-0 input contribution, f32)
  gemm_bf16_kernel<false, false><<<dim3(G4 / 64, 1), 256, 0, stream>>>(
      aenc, Wb_ih0, nullptr, x0buf, HH, G4);

  // 4) 127 sequential steps, 2 fused kernels each (stream-ordered)
  for (int t = 0; t < NSTEPS; ++t) {
    int p = t & 1;
    bf16_t* h0p = h0 + (size_t)p * BB * HH;
    bf16_t* h0n = h0 + (size_t)(p ^ 1) * BB * HH;
    bf16_t* h1p = h1 + (size_t)p * BB * HH;
    bf16_t* h1n = h1 + (size_t)(p ^ 1) * BB * HH;
    lstm_l0_step_kernel<<<64, 256, 0, stream>>>(h0p, h0n, c0, Wb_hh0, W_ih0T,
                                                x0buf, text, t, b_ih0, b_hh0);
    lstm_l1_step_kernel<<<64, 256, 0, stream>>>(h0n, h1p, h1n, c1, Wb_ih1,
                                                Wb_hh1, b_ih1, b_hh1,
                                                hs + (size_t)t * BB * HH);
  }

  // 5) logits = hs @ W_out^T + b_out  (M=8128, f32 out)
  gemm_bf16_kernel<true, false><<<dim3(HH / 64, NSTEPS), 256, 0, stream>>>(
      hs, Wb_out, b_out, logits, HH, HH);
}
